// GAT_GCN_64476049047831
// MI455X (gfx1250) — matrix operand order repaired and checked by reading, not compiled, not run on hardware
//
#include <hip/hip_runtime.h>
#include <hip/hip_bf16.h>
#include <math.h>

#define N_NODES 8000
#define N_EDGES 32000
#define N_GRAPH 256
#define F_IN    680
#define N_HEAD  10
#define HF      6800          // F_IN * N_HEAD
#define ETOT    (N_EDGES + N_NODES)
#define KP1     704           // F_IN padded to multiple of 32
#define KP2     6816          // HF padded to multiple of 32
#define H2F     13600         // 2*HF (already multiple of 32)
#define MP      8064          // N_NODES padded to multiple of 128 (A-row pad)
#define NP      6912          // HF padded to multiple of 128 (B-col pad)

typedef __attribute__((ext_vector_type(16))) __bf16 v16bf;
typedef __attribute__((ext_vector_type(8)))  float  v8f;

// ---------------------------------------------------------------------------
// float atomic max via int/uint atomics (valid for all finite + -inf values)
// ---------------------------------------------------------------------------
__device__ __forceinline__ void atomicMaxF(float* addr, float val) {
    if (val >= 0.0f) atomicMax(reinterpret_cast<int*>(addr), __float_as_int(val));
    else             atomicMin(reinterpret_cast<unsigned int*>(addr), __float_as_uint(val));
}

__device__ __forceinline__ void edge_sd(const int* ei, int e, int& s, int& d) {
    if (e < N_EDGES) { s = ei[e]; d = ei[N_EDGES + e]; }
    else             { s = e - N_EDGES; d = s; }          // self loops appended
}

// ---------------------------------------------------------------------------
// fp32 -> bf16 conversion with zero padding: in[R,C] -> out[Rp,Cp]
// ---------------------------------------------------------------------------
__global__ void cvt_bf16_kernel(const float* __restrict__ in, __bf16* __restrict__ out,
                                int R, int C, int Rp, int Cp) {
    size_t i = (size_t)blockIdx.x * blockDim.x + threadIdx.x;
    size_t total = (size_t)Rp * Cp;
    if (i >= total) return;
    int r = (int)(i / Cp);
    int c = (int)(i - (size_t)r * Cp);
    out[i] = (r < R && c < C) ? (__bf16)in[(size_t)r * C + c] : (__bf16)0.0f;
}

__global__ void fill_kernel(float* __restrict__ p, float v, size_t n) {
    size_t i = (size_t)blockIdx.x * blockDim.x + threadIdx.x;
    if (i < n) p[i] = v;
}

// ---------------------------------------------------------------------------
// Tiled WMMA GEMM:  C[M,N](f32) = A[M,K](bf16) @ B[K,N](bf16)  (+bias, relu)
//
// Requirements (guaranteed by callers via padded staging buffers):
//   - K multiple of 32
//   - A allocated with rows >= ceil(M/128)*128, zero-padded
//   - B allocated with ldb cols covering ceil(N/128)*128, zero-padded
// Block: 256 thr = 8 waves.  Block tile 128x128, wave tile 32x64 (2x4 WMMA).
// Register double-buffering: next K-tile is prefetched into VGPRs while the
// current tile's WMMAs run, then committed to LDS after the second barrier.
// ---------------------------------------------------------------------------
__global__ __launch_bounds__(256)
void gemm_bf16_wmma(const __bf16* __restrict__ A, const __bf16* __restrict__ B,
                    float* __restrict__ C, int M, int N, int K,
                    int lda, int ldb, int ldc,
                    const float* __restrict__ bias, int relu)
{
    __shared__ alignas(16) __bf16 As[128 * 40];   // [row][k], stride 40 (80B, 16B-aligned)
    __shared__ alignas(16) __bf16 Bs[128 * 40];   // transposed [n][k], stride 40

    const int t    = threadIdx.x;
    const int lane = t & 31;
    const int wave = t >> 5;        // 0..7
    const int wm   = wave & 3;      // 4 row groups of 32
    const int wn   = wave >> 2;     // 2 col groups of 64
    const int m0   = blockIdx.y * 128;
    const int n0   = blockIdx.x * 128;

    v8f acc[2][4] = {};

    // global->LDS staging mapping (all loads unconditional: buffers padded)
    const int arow = t >> 1;            // 0..127
    const int acol = (t & 1) * 16;      // 0 or 16
    const int bk   = t >> 3;            // 0..31
    const int bn   = (t & 7) * 8;       // 0..56 (plus +64 for second half)

    const __bf16* ap = A + (size_t)(m0 + arow) * lda + acol;
    const __bf16* bp = B + (size_t)bk * ldb + n0 + bn;

    union B8 { uint4 u; __bf16 h[8]; };
    uint4 a0 = *reinterpret_cast<const uint4*>(ap);
    uint4 a1 = *reinterpret_cast<const uint4*>(ap + 8);
    B8 b0, b1;
    b0.u = *reinterpret_cast<const uint4*>(bp);
    b1.u = *reinterpret_cast<const uint4*>(bp + 64);

    // fragment addressing per ISA 16-bit layouts
    const int am  = lane & 15;
    const int akb = (lane < 16) ? 0 : 8;    // A: K runs {akb..akb+7, akb+16..akb+23}
    const int bnl = lane & 15;
    const int bkb = (lane < 16) ? 0 : 8;    // B: K runs {bkb..bkb+7, bkb+16..bkb+23}

    int kk = 0;
    for (;;) {
        // ---- commit staged K-tile to LDS
        {
            uint4* s = reinterpret_cast<uint4*>(As + arow * 40 + acol);
            s[0] = a0; s[1] = a1;
#pragma unroll
            for (int i = 0; i < 8; i++) {
                Bs[(bn + i) * 40 + bk]      = b0.h[i];
                Bs[(bn + 64 + i) * 40 + bk] = b1.h[i];
            }
        }
        __syncthreads();

        kk += 32;
        const bool more = (kk < K);
        if (more) {  // prefetch next K-tile into registers while WMMAs run
            ap += 32;
            bp += (size_t)32 * ldb;
            a0 = *reinterpret_cast<const uint4*>(ap);
            a1 = *reinterpret_cast<const uint4*>(ap + 8);
            b0.u = *reinterpret_cast<const uint4*>(bp);
            b1.u = *reinterpret_cast<const uint4*>(bp + 64);
        }

        // ---- fragments + 2x4 WMMA
        union AB { uint4 u[2]; v16bf v; };
        AB afrag[2], bfrag[4];
#pragma unroll
        for (int i = 0; i < 2; i++) {
            const __bf16* p = As + (wm * 32 + i * 16 + am) * 40 + akb;
            afrag[i].u[0] = *reinterpret_cast<const uint4*>(p);
            afrag[i].u[1] = *reinterpret_cast<const uint4*>(p + 16);
        }
#pragma unroll
        for (int j = 0; j < 4; j++) {
            const __bf16* p = Bs + (wn * 64 + j * 16 + bnl) * 40 + bkb;
            bfrag[j].u[0] = *reinterpret_cast<const uint4*>(p);
            bfrag[j].u[1] = *reinterpret_cast<const uint4*>(p + 16);
        }
#pragma unroll
        for (int i = 0; i < 2; i++)
#pragma unroll
            for (int j = 0; j < 4; j++)
                acc[i][j] = __builtin_amdgcn_wmma_f32_16x16x32_bf16(
                    false, afrag[i].v, false, bfrag[j].v, (short)0, acc[i][j], false, false);

        if (!more) break;
        __syncthreads();
    }

    // ---- store C (lane: n = l%16, m = vgpr + (l<16?0:8))
    const int cn  = lane & 15;
    const int mof = (lane < 16) ? 0 : 8;
#pragma unroll
    for (int i = 0; i < 2; i++) {
#pragma unroll
        for (int j = 0; j < 4; j++) {
            int col = n0 + wn * 64 + j * 16 + cn;
            if (col >= N) continue;
            float bv = bias ? bias[col] : 0.0f;
#pragma unroll
            for (int r = 0; r < 8; r++) {
                int row = m0 + wm * 32 + i * 16 + mof + r;
                if (row < M) {
                    float v = acc[i][j][r] + bv;
                    if (relu) v = v > 0.0f ? v : 0.0f;
                    C[(size_t)row * ldc + col] = v;
                }
            }
        }
    }
}

// ---------------------------------------------------------------------------
// attention logits: one wave per (node, head); dot of length F_IN
// ---------------------------------------------------------------------------
__global__ __launch_bounds__(256)
void attn_logits_kernel(const float* __restrict__ xh,
                        const float* __restrict__ a_src, const float* __restrict__ a_dst,
                        float* __restrict__ al_s, float* __restrict__ al_d)
{
    int wid  = (int)(((size_t)blockIdx.x * blockDim.x + threadIdx.x) >> 5);
    int lane = threadIdx.x & 31;
    if (wid >= N_NODES * N_HEAD) return;
    int n = wid / N_HEAD, h = wid - n * N_HEAD;
    const float* v  = xh + (size_t)n * HF + h * F_IN;
    const float* as = a_src + h * F_IN;
    const float* ad = a_dst + h * F_IN;
    float ss = 0.0f, sd = 0.0f;
    for (int f = lane; f < F_IN; f += 32) { float xv = v[f]; ss += xv * as[f]; sd += xv * ad[f]; }
    for (int off = 16; off > 0; off >>= 1) {
        ss += __shfl_down(ss, off, 32);
        sd += __shfl_down(sd, off, 32);
    }
    if (lane == 0) { al_s[wid] = ss; al_d[wid] = sd; }
}

// ---------------------------------------------------------------------------
// edge passes for segment softmax over dst
// ---------------------------------------------------------------------------
__global__ void edge_max_kernel(const int* __restrict__ ei,
                                const float* __restrict__ al_s, const float* __restrict__ al_d,
                                float* __restrict__ m)
{
    int idx = blockIdx.x * blockDim.x + threadIdx.x;
    if (idx >= ETOT * N_HEAD) return;
    int e = idx / N_HEAD, h = idx - e * N_HEAD;
    int s, d; edge_sd(ei, e, s, d);
    float v = al_s[s * N_HEAD + h] + al_d[d * N_HEAD + h];
    v = v > 0.0f ? v : 0.2f * v;                        // leaky relu 0.2
    atomicMaxF(&m[d * N_HEAD + h], v);
}

__global__ void edge_exp_kernel(const int* __restrict__ ei,
                                const float* __restrict__ al_s, const float* __restrict__ al_d,
                                const float* __restrict__ m,
                                float* __restrict__ ex, float* __restrict__ z)
{
    int idx = blockIdx.x * blockDim.x + threadIdx.x;
    if (idx >= ETOT * N_HEAD) return;
    int e = idx / N_HEAD, h = idx - e * N_HEAD;
    int s, d; edge_sd(ei, e, s, d);
    float v = al_s[s * N_HEAD + h] + al_d[d * N_HEAD + h];
    v = v > 0.0f ? v : 0.2f * v;
    float exv = expf(v - m[d * N_HEAD + h]);
    ex[idx] = exv;
    atomicAdd(&z[d * N_HEAD + h], exv);
}

// ---------------------------------------------------------------------------
// GAT aggregation: out[dst] += alpha * xh[src]   (one block per edge)
// ---------------------------------------------------------------------------
__global__ __launch_bounds__(256)
void gat_aggregate_kernel(const int* __restrict__ ei, const float* __restrict__ xh,
                          const float* __restrict__ ex, const float* __restrict__ z,
                          float* __restrict__ out)
{
    int e = blockIdx.x;
    int s, d; edge_sd(ei, e, s, d);
    __shared__ float alpha[N_HEAD];
    if (threadIdx.x < N_HEAD)
        alpha[threadIdx.x] = ex[e * N_HEAD + threadIdx.x] /
                             (z[d * N_HEAD + threadIdx.x] + 1e-16f);
    __syncthreads();
    const float* xs = xh + (size_t)s * HF;
    float* od = out + (size_t)d * HF;
    for (int c = threadIdx.x; c < HF; c += 256) {
        int h = c / F_IN;
        atomicAdd(&od[c], alpha[h] * xs[c]);
    }
}

__global__ void bias_act_kernel(float* __restrict__ buf, const float* __restrict__ bias,
                                size_t total, int cols, int relu)
{
    size_t i = (size_t)blockIdx.x * blockDim.x + threadIdx.x;
    if (i >= total) return;
    int c = (int)(i % (size_t)cols);
    float v = buf[i] + bias[c];
    if (relu) v = v > 0.0f ? v : 0.0f;
    buf[i] = v;
}

__global__ void deg_kernel(const int* __restrict__ ei, float* __restrict__ deg) {
    int e = blockIdx.x * blockDim.x + threadIdx.x;
    if (e >= ETOT) return;
    int s, d; edge_sd(ei, e, s, d);
    atomicAdd(&deg[d], 1.0f);
}

__global__ void rsqrt_kernel(float* __restrict__ p, int n) {
    int i = blockIdx.x * blockDim.x + threadIdx.x;
    if (i < n) p[i] = rsqrtf(p[i]);
}

__global__ __launch_bounds__(256)
void gcn_aggregate_kernel(const int* __restrict__ ei, const float* __restrict__ xw,
                          const float* __restrict__ dinv, float* __restrict__ out)
{
    int e = blockIdx.x;
    int s, d; edge_sd(ei, e, s, d);
    float norm = dinv[s] * dinv[d];
    const float* xs = xw + (size_t)s * HF;
    float* od = out + (size_t)d * HF;
    for (int c = threadIdx.x; c < HF; c += 256)
        atomicAdd(&od[c], norm * xs[c]);
}

// ---------------------------------------------------------------------------
// pooling: per-node block does atomic max + add into graph buffers
// ---------------------------------------------------------------------------
__global__ void cnt_kernel(const int* __restrict__ batch, float* __restrict__ cnt) {
    int n = blockIdx.x * blockDim.x + threadIdx.x;
    if (n >= N_NODES) return;
    int b = batch[n];
    if (b < N_GRAPH) atomicAdd(&cnt[b], 1.0f);   // OOB batch ids dropped (jax semantics)
}

__global__ __launch_bounds__(256)
void pool_kernel(const int* __restrict__ batch, const float* __restrict__ h,
                 float* __restrict__ gmax, float* __restrict__ gsum)
{
    int n = blockIdx.x;
    int b = batch[n];
    if (b >= N_GRAPH) return;
    const float* hn = h + (size_t)n * HF;
    float* mx = gmax + (size_t)b * HF;
    float* sm = gsum + (size_t)b * HF;
    for (int c = threadIdx.x; c < HF; c += 256) {
        float v = hn[c];
        atomicMaxF(&mx[c], v);
        atomicAdd(&sm[c], v);
    }
}

__global__ void pool_finalize_kernel(const float* __restrict__ gmax, const float* __restrict__ gsum,
                                     const float* __restrict__ cnt, __bf16* __restrict__ gcat)
{
    int i = blockIdx.x * blockDim.x + threadIdx.x;
    if (i >= N_GRAPH * H2F) return;
    int g = i / H2F, c = i - g * H2F;
    float v = (c < HF) ? gmax[(size_t)g * HF + c]
                       : gsum[(size_t)g * HF + (c - HF)] / cnt[g];
    gcat[i] = (__bf16)v;
}

// ---------------------------------------------------------------------------
// small fp32 dense layers (256x512x128, 256x128x64)
// ---------------------------------------------------------------------------
__global__ void dense_kernel(const float* __restrict__ A, const float* __restrict__ W,
                             const float* __restrict__ b, float* __restrict__ out,
                             int M, int K, int Nn, int relu)
{
    int i = blockIdx.x * blockDim.x + threadIdx.x;
    if (i >= M * Nn) return;
    int r = i / Nn, c = i - r * Nn;
    float acc = b[c];
    const float* ar = A + (size_t)r * K;
    for (int k = 0; k < K; k++) acc += ar[k] * W[(size_t)k * Nn + c];
    if (relu) acc = acc > 0.0f ? acc : 0.0f;
    out[i] = acc;
}

// ---------------------------------------------------------------------------
extern "C" void kernel_launch(void* const* d_in, const int* in_sizes, int n_in,
                              void* d_out, int out_size, void* d_ws, size_t ws_size,
                              hipStream_t stream)
{
    (void)in_sizes; (void)n_in; (void)out_size; (void)ws_size;

    const float* x     = (const float*)d_in[0];
    const int*   ei    = (const int*)  d_in[1];
    const int*   batch = (const int*)  d_in[2];
    const float* Wgat  = (const float*)d_in[3];
    const float* a_src = (const float*)d_in[4];
    const float* a_dst = (const float*)d_in[5];
    const float* b_gat = (const float*)d_in[6];
    const float* Wgcn  = (const float*)d_in[7];
    const float* b_gcn = (const float*)d_in[8];
    const float* W1    = (const float*)d_in[9];
    const float* b1    = (const float*)d_in[10];
    const float* W2    = (const float*)d_in[11];
    const float* b2    = (const float*)d_in[12];
    const float* W3    = (const float*)d_in[13];
    const float* b3    = (const float*)d_in[14];
    float* out = (float*)d_out;

    // ---- workspace bump allocator (256B aligned) ----
    char* base = (char*)d_ws;
    size_t off = 0;
    auto alloc = [&](size_t bytes) -> void* {
        void* p = base + off;
        off = (off + bytes + 255) & ~(size_t)255;
        return p;
    };
    __bf16* xbf  = (__bf16*)alloc((size_t)MP * KP1 * 2);       // A pad: rows 8064
    __bf16* wgat = (__bf16*)alloc((size_t)KP1 * NP * 2);       // B pad: cols 6912
    float*  xh   = (float*) alloc((size_t)N_NODES * HF * 4);   // xh, later reused as xw
    float*  hbuf = (float*) alloc((size_t)N_NODES * HF * 4);   // GAT out -> h -> GCN out -> h2
    float*  al_s = (float*) alloc((size_t)N_NODES * N_HEAD * 4);
    float*  al_d = (float*) alloc((size_t)N_NODES * N_HEAD * 4);
    float*  mbuf = (float*) alloc((size_t)N_NODES * N_HEAD * 4);
    float*  zbuf = (float*) alloc((size_t)N_NODES * N_HEAD * 4);
    float*  exbf = (float*) alloc((size_t)ETOT * N_HEAD * 4);
    float*  deg  = (float*) alloc((size_t)N_NODES * 4);
    __bf16* hbf  = (__bf16*)alloc((size_t)MP * KP2 * 2);
    __bf16* wgcn = (__bf16*)alloc((size_t)KP2 * NP * 2);
    float*  gmax = (float*) alloc((size_t)N_GRAPH * HF * 4);
    float*  gsum = (float*) alloc((size_t)N_GRAPH * HF * 4);
    float*  cnt  = (float*) alloc((size_t)N_GRAPH * 4);
    __bf16* gcat = (__bf16*)alloc((size_t)N_GRAPH * H2F * 2);  // 256 rows = 2*128 exact
    __bf16* w1bf = (__bf16*)alloc((size_t)H2F * 512 * 2);      // 512 cols = 4*128 exact
    float*  y1   = (float*) alloc((size_t)N_GRAPH * 512 * 4);
    float*  y2   = (float*) alloc((size_t)N_GRAPH * 128 * 4);

    auto cdiv = [](size_t a, size_t b) { return (unsigned)((a + b - 1) / b); };
    const size_t NH  = (size_t)N_NODES * N_HEAD;
    const size_t NHF = (size_t)N_NODES * HF;

    // ---- 1. GAT linear: xh = x @ W_gat (bf16 WMMA; K 680->704, M 8000->8064, N 6800->6912)
    cvt_bf16_kernel<<<cdiv((size_t)MP * KP1, 256), 256, 0, stream>>>(x, xbf, N_NODES, F_IN, MP, KP1);
    cvt_bf16_kernel<<<cdiv((size_t)KP1 * NP, 256), 256, 0, stream>>>(Wgat, wgat, F_IN, HF, KP1, NP);
    gemm_bf16_wmma<<<dim3(cdiv(HF, 128), cdiv(N_NODES, 128)), 256, 0, stream>>>(
        xbf, wgat, xh, N_NODES, HF, KP1, KP1, NP, HF, nullptr, 0);

    // ---- 2. attention logits + segment softmax over dst
    attn_logits_kernel<<<cdiv(NH * 32, 256), 256, 0, stream>>>(xh, a_src, a_dst, al_s, al_d);
    fill_kernel<<<cdiv(NH, 256), 256, 0, stream>>>(mbuf, -INFINITY, NH);
    fill_kernel<<<cdiv(NH, 256), 256, 0, stream>>>(zbuf, 0.0f, NH);
    fill_kernel<<<cdiv(NHF, 256), 256, 0, stream>>>(hbuf, 0.0f, NHF);
    fill_kernel<<<cdiv((size_t)N_NODES, 256), 256, 0, stream>>>(deg, 0.0f, N_NODES);
    edge_max_kernel<<<cdiv((size_t)ETOT * N_HEAD, 256), 256, 0, stream>>>(ei, al_s, al_d, mbuf);
    edge_exp_kernel<<<cdiv((size_t)ETOT * N_HEAD, 256), 256, 0, stream>>>(ei, al_s, al_d, mbuf, exbf, zbuf);

    // ---- 3. GAT aggregation + bias + relu  -> h in hbuf
    gat_aggregate_kernel<<<ETOT, 256, 0, stream>>>(ei, xh, exbf, zbuf, hbuf);
    bias_act_kernel<<<cdiv(NHF, 256), 256, 0, stream>>>(hbuf, b_gat, NHF, HF, 1);

    // ---- 4. GCN degrees
    deg_kernel<<<cdiv((size_t)ETOT, 256), 256, 0, stream>>>(ei, deg);
    rsqrt_kernel<<<cdiv((size_t)N_NODES, 256), 256, 0, stream>>>(deg, N_NODES);   // deg -> dinv

    // ---- 5. GCN linear: xw = h @ W_gcn (bf16 WMMA; K 6800->6816), xw reuses xh buffer
    cvt_bf16_kernel<<<cdiv((size_t)MP * KP2, 256), 256, 0, stream>>>(hbuf, hbf, N_NODES, HF, MP, KP2);
    cvt_bf16_kernel<<<cdiv((size_t)KP2 * NP, 256), 256, 0, stream>>>(Wgcn, wgcn, HF, HF, KP2, NP);
    gemm_bf16_wmma<<<dim3(cdiv(HF, 128), cdiv(N_NODES, 128)), 256, 0, stream>>>(
        hbf, wgcn, xh, N_NODES, HF, KP2, KP2, NP, HF, nullptr, 0);

    // ---- 6. GCN aggregation + bias + relu -> h2 in hbuf (h no longer needed)
    fill_kernel<<<cdiv(NHF, 256), 256, 0, stream>>>(hbuf, 0.0f, NHF);
    gcn_aggregate_kernel<<<ETOT, 256, 0, stream>>>(ei, xh, deg, hbuf);
    bias_act_kernel<<<cdiv(NHF, 256), 256, 0, stream>>>(hbuf, b_gcn, NHF, HF, 1);

    // ---- 7. pooling: global max + mean per graph -> gcat (bf16, [G, 2*HF])
    const size_t GHF = (size_t)N_GRAPH * HF;
    fill_kernel<<<cdiv(GHF, 256), 256, 0, stream>>>(gmax, -INFINITY, GHF);
    fill_kernel<<<cdiv(GHF, 256), 256, 0, stream>>>(gsum, 0.0f, GHF);
    fill_kernel<<<cdiv((size_t)N_GRAPH, 256), 256, 0, stream>>>(cnt, 0.0f, N_GRAPH);
    cnt_kernel<<<cdiv((size_t)N_NODES, 256), 256, 0, stream>>>(batch, cnt);
    pool_kernel<<<N_NODES, 256, 0, stream>>>(batch, hbuf, gmax, gsum);
    pool_finalize_kernel<<<cdiv((size_t)N_GRAPH * H2F, 256), 256, 0, stream>>>(gmax, gsum, cnt, gcat);

    // ---- 8. MLP: y1 = relu(g @ W1 + b1) via WMMA; then small fp32 layers
    cvt_bf16_kernel<<<cdiv((size_t)H2F * 512, 256), 256, 0, stream>>>(W1, w1bf, H2F, 512, H2F, 512);
    gemm_bf16_wmma<<<dim3(cdiv(512, 128), cdiv(N_GRAPH, 128)), 256, 0, stream>>>(
        gcat, w1bf, y1, N_GRAPH, 512, H2F, H2F, 512, 512, b1, 1);
    dense_kernel<<<cdiv((size_t)N_GRAPH * 128, 256), 256, 0, stream>>>(y1, W2, b2, y2, N_GRAPH, 512, 128, 0);
    dense_kernel<<<cdiv((size_t)N_GRAPH * 64, 256), 256, 0, stream>>>(y2, W3, b3, out, N_GRAPH, 128, 64, 0);
}
